// LogicGatedSNN_12695923326984
// MI455X (gfx1250) — matrix operand order repaired and checked by reading, not compiled, not run on hardware
//
#include <hip/hip_runtime.h>
#include <stdint.h>

// ---- types ------------------------------------------------------------
typedef __attribute__((ext_vector_type(16))) __bf16    v16bf;
typedef __attribute__((ext_vector_type(8)))  float     v8f;
typedef __attribute__((ext_vector_type(4)))  float     v4f;
typedef __attribute__((ext_vector_type(4)))  uint32_t  v4u;
typedef __attribute__((ext_vector_type(8)))  uint32_t  v8u;
typedef __attribute__((ext_vector_type(2)))  uint32_t  v2u;

union FragU {          // one WMMA 16-bit operand: 16 bf16 = 8 VGPRs
    v4u   u[2];
    v16bf v;
};

// ---- problem constants (fixed by the reference) -----------------------
#define MDIM 8192
#define NDIM 4096
#define KDIM 4096

#define BM 128
#define BN 128
#define BK 32
#define LSTR 40            // LDS row stride in bf16 elems (BK + 8 pad -> conflict-free b128)
#define THREADS 512        // 16 waves (wave32), 4x4 wave grid, 32x32 per wave

// LDS layout (bytes): per buffer {Ahi, Alo, W}, each BM*LSTR*2 = 10240 B
#define ARR_B   (BM * LSTR * 2)        // 10240
#define BUF_B   (3 * ARR_B)            // 30720
#define LDS_B   (2 * BUF_B)            // 61440
#define ARR_E   (BM * LSTR)            // elems
#define BUF_E   (3 * ARR_E)

// round-to-nearest-even fp32 -> bf16 (bits)
__device__ __forceinline__ uint32_t bf16_rn(float f) {
    uint32_t u = __float_as_uint(f);
    return (u + 0x7FFFu + ((u >> 16) & 1u)) >> 16;
}

// split 4 floats into hi/lo bf16 pairs (x == hi + lo to 16 mantissa bits)
__device__ __forceinline__ void cvt_hilo4(const v4f a, v2u& hi, v2u& lo) {
    uint32_t h0 = bf16_rn(a.x), h1 = bf16_rn(a.y), h2 = bf16_rn(a.z), h3 = bf16_rn(a.w);
    float r0 = a.x - __uint_as_float(h0 << 16);
    float r1 = a.y - __uint_as_float(h1 << 16);
    float r2 = a.z - __uint_as_float(h2 << 16);
    float r3 = a.w - __uint_as_float(h3 << 16);
    uint32_t l0 = bf16_rn(r0), l1 = bf16_rn(r1), l2 = bf16_rn(r2), l3 = bf16_rn(r3);
    hi.x = h0 | (h1 << 16);  hi.y = h2 | (h3 << 16);
    lo.x = l0 | (l1 << 16);  lo.y = l2 | (l3 << 16);
}

// ternarize one state -> bf16 bits of {-1, 0, +1}
__device__ __forceinline__ uint32_t tern1(float s) {
    return s > 1.0f ? 0x3F80u : (s < -1.0f ? 0xBF80u : 0u);
}
__device__ __forceinline__ v2u tern4(const v4f s) {
    v2u r;
    r.x = tern1(s.x) | (tern1(s.y) << 16);
    r.y = tern1(s.z) | (tern1(s.w) << 16);
    return r;
}

// =======================================================================
// Prep kernels (memory-bound, run once per launch): fp32 -> bf16 planes
// =======================================================================
__global__ __launch_bounds__(256)
void prep_split_x(const float* __restrict__ X, uint16_t* __restrict__ Ahi,
                  uint16_t* __restrict__ Alo) {
    size_t i = ((size_t)blockIdx.x * blockDim.x + threadIdx.x) * 4;
    v4f a = *(const v4f*)(X + i);
    v2u hi, lo;
    cvt_hilo4(a, hi, lo);
    *(v2u*)(Ahi + i) = hi;
    *(v2u*)(Alo + i) = lo;
}

__global__ __launch_bounds__(256)
void prep_tern_w(const float* __restrict__ S, uint16_t* __restrict__ Wt) {
    size_t i = ((size_t)blockIdx.x * blockDim.x + threadIdx.x) * 4;
    v4f s = *(const v4f*)(S + i);
    *(v2u*)(Wt + i) = tern4(s);
}

// =======================================================================
// TDM: issue a 2D tile DMA (tile BK x BM bf16, padded rows) into LDS.
// D# built per CDNA5 ISA ch.8: group0 (4 SGPRs) + group1 (8 SGPRs),
// groups 2/3 omitted (<=2D). pad: every 16 DWORDs (64B row) add 4 DWORDs
// (16B) -> LDS row stride 80B == LSTR bf16 elems.
// =======================================================================
__device__ __forceinline__ void tdm_load_tile_2d(uint32_t lds_byte_off,
                                                 const uint16_t* gptr,
                                                 uint32_t tensor_rows) {
    uint64_t ga = (uint64_t)(uintptr_t)gptr;
    v4u g0;
    g0[0] = 0x1u;                                   // count=1, user mode, no gather
    g0[1] = lds_byte_off;                           // lds_addr [63:32]
    g0[2] = (uint32_t)ga;                           // global_addr lo
    g0[3] = (uint32_t)(ga >> 32) | (2u << 30);      // global_addr hi | type=2 (image)
    v8u g1;
    g1[0] = (1u << 16)                              // data_size = 2 bytes
          | (1u << 20)                              // pad_enable
          | (3u << 22)                              // pad_interval = 16 DWORDs (64B)
          | (3u << 25);                             // pad_amount   =  4 DWORDs (16B)
    g1[1] = ((uint32_t)KDIM & 0xFFFFu) << 16;       // tensor_dim0 lo16 (bits 79:48)
    g1[2] = ((uint32_t)KDIM >> 16)                  // tensor_dim0 hi16
          | ((tensor_rows & 0xFFFFu) << 16);        // tensor_dim1 lo16
    g1[3] = (tensor_rows >> 16)                     // tensor_dim1 hi16
          | ((uint32_t)BK << 16);                   // tile_dim0 = 32
    g1[4] = (uint32_t)BM;                           // tile_dim1 = 128, tile_dim2 = 0
    g1[5] = (uint32_t)KDIM;                         // tensor_dim0_stride lo32
    g1[6] = 0;                                      // stride hi16 | dim1_stride lo16
    g1[7] = 0;                                      // dim1_stride hi32
    asm volatile("tensor_load_to_lds %0, %1" :: "s"(g0), "s"(g1) : "memory");
}

// =======================================================================
// Main GEMM: bf16 planes from d_ws, TDM-staged LDS, double buffered.
// Inner loop per wave: 12x ds_load_b128 + 8x v_wmma_f32_16x16x32_bf16.
// =======================================================================
__global__ __launch_bounds__(THREADS, 1)
void snn_gemm_wmma_tdm(const uint16_t* __restrict__ Ahi,
                       const uint16_t* __restrict__ Alo,
                       const uint16_t* __restrict__ Wt,
                       float* __restrict__ Out) {
    extern __shared__ uint16_t smem[];   // dynamic LDS: offset 0 of group segment

    const int tid  = threadIdx.x;
    const int bm   = blockIdx.y * BM;
    const int bn   = blockIdx.x * BN;
    const int wave = tid >> 5;
    const int lane = tid & 31;
    const int wm   = wave >> 2;          // 0..3
    const int wn   = wave & 3;           // 0..3
    const int r    = lane & 15;
    const int h    = lane >> 4;

    // A frag (16x32 MxK): lanes 0-15 K 0..7 & 16..23; lanes 16-31 K 8..15 & 24..31
    int aOff[2], bOff[2];
    aOff[0] = (wm * 32 +  0 + r) * LSTR + h * 8;
    aOff[1] = (wm * 32 + 16 + r) * LSTR + h * 8;
    // B frag (32x16 KxN): lanes 0-15 K 0..7 & 16..23; lanes 16-31 K 8..15 & 24..31
    bOff[0] = (wn * 32 +  0 + r) * LSTR + h * 8;
    bOff[1] = (wn * 32 + 16 + r) * LSTR + h * 8;

    v8f acc[2][2];
    #pragma unroll
    for (int mi = 0; mi < 2; ++mi)
        #pragma unroll
        for (int ni = 0; ni < 2; ++ni)
            acc[mi][ni] = (v8f)(0.0f);

    const uint16_t* Ag = Ahi + (size_t)bm * KDIM;
    const uint16_t* Lg = Alo + (size_t)bm * KDIM;
    const uint16_t* Wg = Wt  + (size_t)bn * KDIM;

    // stage 0
    if (wave == 0) {
        tdm_load_tile_2d(0 * BUF_B + 0 * ARR_B, Ag, MDIM);
        tdm_load_tile_2d(0 * BUF_B + 1 * ARR_B, Lg, MDIM);
        tdm_load_tile_2d(0 * BUF_B + 2 * ARR_B, Wg, NDIM);
    }
    __builtin_amdgcn_s_wait_tensorcnt(0);
    __syncthreads();

    const int NK = KDIM / BK;            // 128 K-steps
    for (int kt = 0; kt < NK; ++kt) {
        const int cur = kt & 1;

        // kick off DMA of the next K-slice into the other buffer
        if (kt + 1 < NK && wave == 0) {
            const int nxt = cur ^ 1;
            const size_t ko = (size_t)(kt + 1) * BK;
            tdm_load_tile_2d(nxt * BUF_B + 0 * ARR_B, Ag + ko, MDIM);
            tdm_load_tile_2d(nxt * BUF_B + 1 * ARR_B, Lg + ko, MDIM);
            tdm_load_tile_2d(nxt * BUF_B + 2 * ARR_B, Wg + ko, NDIM);
        }

        // fragment loads (ds_load_b128 pairs) from current buffer
        const uint16_t* sA = smem + cur * BUF_E;
        const uint16_t* sL = sA + ARR_E;
        const uint16_t* sB = sA + 2 * ARR_E;
        FragU ah[2], al[2], bw[2];
        #pragma unroll
        for (int i = 0; i < 2; ++i) {
            const uint16_t* pa = sA + aOff[i];
            ah[i].u[0] = *(const v4u*)(pa);
            ah[i].u[1] = *(const v4u*)(pa + 16);
            const uint16_t* pl = sL + aOff[i];
            al[i].u[0] = *(const v4u*)(pl);
            al[i].u[1] = *(const v4u*)(pl + 16);
            const uint16_t* pb = sB + bOff[i];
            bw[i].u[0] = *(const v4u*)(pb);
            bw[i].u[1] = *(const v4u*)(pb + 16);
        }

        #pragma unroll
        for (int mi = 0; mi < 2; ++mi)
            #pragma unroll
            for (int ni = 0; ni < 2; ++ni) {
                acc[mi][ni] = __builtin_amdgcn_wmma_f32_16x16x32_bf16(
                    false, ah[mi].v, false, bw[ni].v, (short)0, acc[mi][ni], false, false);
                acc[mi][ni] = __builtin_amdgcn_wmma_f32_16x16x32_bf16(
                    false, al[mi].v, false, bw[ni].v, (short)0, acc[mi][ni], false, false);
            }

        __builtin_amdgcn_s_wait_tensorcnt(0);   // no-op for non-issuing waves
        __syncthreads();
    }

    // epilogue: spikes = (acc >= 1.0). C/D layout: VGPR j -> row h*8+j, col r
    #pragma unroll
    for (int mi = 0; mi < 2; ++mi) {
        #pragma unroll
        for (int ni = 0; ni < 2; ++ni) {
            const int rowBase = bm + wm * 32 + mi * 16 + h * 8;
            const int col     = bn + wn * 32 + ni * 16 + r;
            #pragma unroll
            for (int j = 0; j < 8; ++j) {
                float v = acc[mi][ni][j] >= 1.0f ? 1.0f : 0.0f;
                Out[(size_t)(rowBase + j) * NDIM + col] = v;
            }
        }
    }
}

// =======================================================================
// Fallback (ws too small): round-1 fused kernel (convert-in-loop).
// =======================================================================
__global__ __launch_bounds__(THREADS, 1)
void snn_ternary_gemm_fused(const float* __restrict__ X,
                            const float* __restrict__ S,
                            float* __restrict__ Out) {
    __shared__ uint16_t sAhi[2][BM * LSTR];
    __shared__ uint16_t sAlo[2][BM * LSTR];
    __shared__ uint16_t sW  [2][BM * LSTR];

    const int tid = threadIdx.x;
    const int bm  = blockIdx.y * BM;
    const int bn  = blockIdx.x * BN;

    const int chunk = tid & 7;
    const int row0  = tid >> 3;

    const float* Ag0 = X + (size_t)(bm + row0) * KDIM + chunk * 4;
    const float* Ag1 = Ag0 + (size_t)64 * KDIM;
    const float* Wg0 = S + (size_t)(bn + row0) * KDIM + chunk * 4;
    const float* Wg1 = Wg0 + (size_t)64 * KDIM;

    const int stOff0 = row0 * LSTR + chunk * 4;
    const int stOff1 = (row0 + 64) * LSTR + chunk * 4;

    const int wave = tid >> 5;
    const int lane = tid & 31;
    const int wm = wave >> 2;
    const int wn = wave & 3;
    const int r  = lane & 15;
    const int h  = lane >> 4;

    int aOff[2], bOff[2];
    aOff[0] = (wm * 32 +  0 + r) * LSTR + h * 8;
    aOff[1] = (wm * 32 + 16 + r) * LSTR + h * 8;
    bOff[0] = (wn * 32 +  0 + r) * LSTR + h * 8;
    bOff[1] = (wn * 32 + 16 + r) * LSTR + h * 8;

    v8f acc[2][2];
    #pragma unroll
    for (int mi = 0; mi < 2; ++mi)
        #pragma unroll
        for (int ni = 0; ni < 2; ++ni)
            acc[mi][ni] = (v8f)(0.0f);

    v4f pa0 = *(const v4f*)(Ag0);
    v4f pa1 = *(const v4f*)(Ag1);
    v4f pw0 = *(const v4f*)(Wg0);
    v4f pw1 = *(const v4f*)(Wg1);
    {
        v2u hi, lo;
        cvt_hilo4(pa0, hi, lo);
        *(v2u*)&sAhi[0][stOff0] = hi;  *(v2u*)&sAlo[0][stOff0] = lo;
        cvt_hilo4(pa1, hi, lo);
        *(v2u*)&sAhi[0][stOff1] = hi;  *(v2u*)&sAlo[0][stOff1] = lo;
        *(v2u*)&sW[0][stOff0] = tern4(pw0);
        *(v2u*)&sW[0][stOff1] = tern4(pw1);
    }
    __syncthreads();

    const int NK = KDIM / BK;
    for (int kt = 0; kt < NK; ++kt) {
        const int cur = kt & 1;
        if (kt + 1 < NK) {
            const size_t koff = (size_t)(kt + 1) * BK;
            pa0 = *(const v4f*)(Ag0 + koff);
            pa1 = *(const v4f*)(Ag1 + koff);
            pw0 = *(const v4f*)(Wg0 + koff);
            pw1 = *(const v4f*)(Wg1 + koff);
        }

        FragU ah[2], al[2], bw[2];
        #pragma unroll
        for (int i = 0; i < 2; ++i) {
            const uint16_t* pa = &sAhi[cur][aOff[i]];
            ah[i].u[0] = *(const v4u*)(pa);
            ah[i].u[1] = *(const v4u*)(pa + 16);
            const uint16_t* pl = &sAlo[cur][aOff[i]];
            al[i].u[0] = *(const v4u*)(pl);
            al[i].u[1] = *(const v4u*)(pl + 16);
            const uint16_t* pb = &sW[cur][bOff[i]];
            bw[i].u[0] = *(const v4u*)(pb);
            bw[i].u[1] = *(const v4u*)(pb + 16);
        }

        #pragma unroll
        for (int mi = 0; mi < 2; ++mi)
            #pragma unroll
            for (int ni = 0; ni < 2; ++ni) {
                acc[mi][ni] = __builtin_amdgcn_wmma_f32_16x16x32_bf16(
                    false, ah[mi].v, false, bw[ni].v, (short)0, acc[mi][ni], false, false);
                acc[mi][ni] = __builtin_amdgcn_wmma_f32_16x16x32_bf16(
                    false, al[mi].v, false, bw[ni].v, (short)0, acc[mi][ni], false, false);
            }

        if (kt + 1 < NK) {
            const int nxt = cur ^ 1;
            v2u hi, lo;
            cvt_hilo4(pa0, hi, lo);
            *(v2u*)&sAhi[nxt][stOff0] = hi;  *(v2u*)&sAlo[nxt][stOff0] = lo;
            cvt_hilo4(pa1, hi, lo);
            *(v2u*)&sAhi[nxt][stOff1] = hi;  *(v2u*)&sAlo[nxt][stOff1] = lo;
            *(v2u*)&sW[nxt][stOff0] = tern4(pw0);
            *(v2u*)&sW[nxt][stOff1] = tern4(pw1);
        }
        __syncthreads();
    }

    #pragma unroll
    for (int mi = 0; mi < 2; ++mi) {
        #pragma unroll
        for (int ni = 0; ni < 2; ++ni) {
            const int rowBase = bm + wm * 32 + mi * 16 + h * 8;
            const int col     = bn + wn * 32 + ni * 16 + r;
            #pragma unroll
            for (int j = 0; j < 8; ++j) {
                float v = acc[mi][ni][j] >= 1.0f ? 1.0f : 0.0f;
                Out[(size_t)(rowBase + j) * NDIM + col] = v;
            }
        }
    }
}

extern "C" void kernel_launch(void* const* d_in, const int* in_sizes, int n_in,
                              void* d_out, int out_size, void* d_ws, size_t ws_size,
                              hipStream_t stream) {
    (void)in_sizes; (void)n_in; (void)out_size;
    const float* X = (const float*)d_in[0];   // spike_input [8192, 4096]
    const float* S = (const float*)d_in[1];   // synapse_states [4096, 4096]
    float* Out = (float*)d_out;               // spikes [8192, 4096]

    const size_t AHI_B = (size_t)MDIM * KDIM * 2;   // 64 MiB
    const size_t WT_B  = (size_t)NDIM * KDIM * 2;   // 32 MiB
    const size_t NEED  = 2 * AHI_B + WT_B;          // 160 MiB

    dim3 grid(NDIM / BN, MDIM / BM);                // (32, 64)
    dim3 block(THREADS);

    if (d_ws != nullptr && ws_size >= NEED) {
        uint16_t* Ahi = (uint16_t*)d_ws;
        uint16_t* Alo = (uint16_t*)((char*)d_ws + AHI_B);
        uint16_t* Wt  = (uint16_t*)((char*)d_ws + 2 * AHI_B);

        prep_split_x<<<(MDIM * (size_t)KDIM) / (4 * 256), 256, 0, stream>>>(X, Ahi, Alo);
        prep_tern_w<<<(NDIM * (size_t)KDIM) / (4 * 256), 256, 0, stream>>>(S, Wt);
        snn_gemm_wmma_tdm<<<grid, block, LDS_B, stream>>>(Ahi, Alo, Wt, Out);
    } else {
        snn_ternary_gemm_fused<<<grid, block, 0, stream>>>(X, S, Out);
    }
}
